// FasterRCNN_63058709840280
// MI455X (gfx1250) — compile-verified, matrix operand order checked by reading, not run on hardware
//
#include <hip/hip_runtime.h>

typedef __attribute__((ext_vector_type(2)))  float    v2f;
typedef __attribute__((ext_vector_type(8)))  float    v8f;
typedef __attribute__((ext_vector_type(8)))  _Float16 v8h;
typedef __attribute__((ext_vector_type(16))) _Float16 v16h;

#define NA 9216      // anchors = 32*32*9
#define TOPN 6000
#define NW 188       // ceil(6000/32)
#define KROI 128
#define FH 32
#define FW 32

// ---------------------------------------------------------------------------
// 3x3 SAME conv + bias (+ReLU), decomposed as 9 shifted GEMMs on
// V_WMMA_F32_16X16X4_F32.  M = cout, N = pixels, K = cin (per tap).
// Boundary handling is loop-invariant per tap: clamped-safe address +
// branchless float mask, so the hot loop is 4 loads + 2 muls + 1 WMMA.
// A 16x4 layout: lanes0-15 rows M, VGPR0=K0,VGPR1=K1; lanes16-31 K2,K3.
// blockDim = 256 (8 waves), wave w handles pixel tile blockIdx.x*8+w.
// ---------------------------------------------------------------------------
__global__ __launch_bounds__(256) void conv3x3_wmma(
    const float* __restrict__ in, const float* __restrict__ wgt,
    const float* __restrict__ bias, float* __restrict__ out,
    int Cin, int Cout, int H, int W, int relu)
{
  const int wave = threadIdx.x >> 5;
  const int lane = threadIdx.x & 31;
  const int lr = lane & 15, half = lane >> 4;
  const int HW = H * W;
  const int p0 = (blockIdx.x * 8 + wave) * 16;
  const int o0 = blockIdx.y * 16;
  if (p0 >= HW) return;                       // wave-uniform
  const int pix = p0 + lr;
  const int wsh = 31 - __clz(W);              // W is a power of two
  const int py = pix >> wsh;
  const int px = pix & (W - 1);
  union { v8f v; float f[8]; } acc;
  #pragma unroll
  for (int r = 0; r < 8; ++r) acc.f[r] = 0.0f;
  union { v2f v; float f[2]; } a, b;
  const int K9 = Cin * 9;
  const float* wrowp = wgt + (size_t)(o0 + lr) * K9;
  const int Cmain = Cin & ~3;
  for (int tap = 0; tap < 9; ++tap) {
    const int dy = tap / 3 - 1, dx = tap - (tap / 3) * 3 - 1;
    const int yy = py + dy, xx = px + dx;
    const bool inb = (yy >= 0) & (yy < H) & (xx >= 0) & (xx < W);
    const float msk = inb ? 1.0f : 0.0f;
    const int yc = min(max(yy, 0), H - 1);    // safe address, masked value
    const int xc = min(max(xx, 0), W - 1);
    const float* bp = in + (size_t)(half * 2) * HW + (size_t)yc * W + xc;
    const float* ap = wrowp + (size_t)(half * 2) * 9 + tap;
    for (int c0 = 0; c0 < Cmain; c0 += 4) {
      a.f[0] = ap[0];
      a.f[1] = ap[9];
      b.f[0] = bp[0] * msk;
      b.f[1] = bp[HW] * msk;
      acc.v = __builtin_amdgcn_wmma_f32_16x16x4_f32(false, a.v, false, b.v,
                                                    (short)0, acc.v, false, false);
      ap += 36;
      bp += (size_t)4 * HW;
    }
    if (Cmain < Cin) {                        // only layer 1 (Cin == 3)
      const int kk = Cmain + half * 2;
      a.f[0] = (kk     < Cin) ? ap[0] : 0.0f;
      a.f[1] = (kk + 1 < Cin) ? ap[9] : 0.0f;
      b.f[0] = (kk     < Cin) ? bp[0]  * msk : 0.0f;
      b.f[1] = (kk + 1 < Cin) ? bp[HW] * msk : 0.0f;
      acc.v = __builtin_amdgcn_wmma_f32_16x16x4_f32(false, a.v, false, b.v,
                                                    (short)0, acc.v, false, false);
    }
  }
  #pragma unroll
  for (int r = 0; r < 8; ++r) {
    int m = o0 + r + (half ? 8 : 0);
    float v = acc.f[r] + bias[m];
    if (relu) v = fmaxf(v, 0.0f);
    out[(size_t)m * HW + p0 + lr] = v;
  }
}

__global__ void maxpool2_k(const float* __restrict__ in, float* __restrict__ out,
                           int C, int H, int W)
{
  int Ho = H >> 1, Wo = W >> 1;
  int n = C * Ho * Wo;
  int i = blockIdx.x * 256 + threadIdx.x;
  if (i >= n) return;
  int c = i / (Ho * Wo);
  int r = i - c * (Ho * Wo);
  int h = r / Wo, w = r - h * Wo;
  const float* p = in + (size_t)c * H * W + (size_t)(2 * h) * W + 2 * w;
  out[i] = fmaxf(fmaxf(p[0], p[1]), fmaxf(p[W], p[W + 1]));
}

__global__ void conv1x1_k(const float* __restrict__ in, const float* __restrict__ w,
                          const float* __restrict__ b, float* __restrict__ out,
                          int Cout, int Cin, int HW)
{
  int t = blockIdx.x * 256 + threadIdx.x;
  if (t >= Cout * HW) return;
  int c = t / HW, p = t - c * HW;
  float s = 0.0f;
  for (int k = 0; k < Cin; ++k) s += in[(size_t)k * HW + p] * w[(size_t)c * Cin + k];
  out[t] = s + b[c];
}

// scores softmax (pairs of channels) + per-anchor max prob
__global__ void scores_k(const float* __restrict__ cls, float* __restrict__ sc2,
                         float* __restrict__ smax)
{
  int i = blockIdx.x * 256 + threadIdx.x;
  if (i >= NA) return;
  int p = i / 9, a = i - (i / 9) * 9;
  float l0 = cls[(size_t)(2 * a) * 1024 + p];
  float l1 = cls[(size_t)(2 * a + 1) * 1024 + p];
  float mx = fmaxf(l0, l1);
  float e0 = expf(l0 - mx), e1 = expf(l1 - mx);
  float s = e0 + e1;
  float p0 = e0 / s, p1 = e1 / s;
  sc2[2 * i] = p0; sc2[2 * i + 1] = p1;
  smax[i] = fmaxf(p0, p1);
}

__global__ void proposals_k(const float* __restrict__ bc, float* __restrict__ prop)
{
  int t = blockIdx.x * 256 + threadIdx.x;
  if (t >= NA * 4) return;
  int i = t >> 2, j = t & 3;
  int p = i / 9, a = i - (i / 9) * 9;
  prop[t] = bc[(size_t)(4 * a + j) * 1024 + p];
}

__global__ void anchors_k(float* __restrict__ anc)
{
  int i = blockIdx.x * 256 + threadIdx.x;
  if (i >= NA) return;
  int p = i / 9, a = i - (i / 9) * 9;
  int ri = a / 3, si = a - ri * 3;
  float ratio = (ri == 0) ? 0.5f : (ri == 1 ? 1.0f : 2.0f);
  float scale = (si == 0) ? 8.0f : (si == 1 ? 16.0f : 32.0f);
  float wr = rintf(sqrtf(256.0f / ratio));     // RNE matches jnp.round
  float hr = rintf(wr * ratio);
  float r0 = 7.0f - 0.5f * (wr - 1.0f), r1 = 7.0f - 0.5f * (hr - 1.0f);
  float r2 = 7.0f + 0.5f * (wr - 1.0f), r3 = 7.0f + 0.5f * (hr - 1.0f);
  // voc_to_yolo coordinate mixing replicated exactly
  float y0 = r0 + 0.5f * (r3 - 1.0f);
  float y1 = r1 + 0.5f * (r2 - 1.0f);
  float y2 = r3 - r1 + 1.0f;
  float y3 = r2 - r0 + 1.0f;
  float ww = y2 * scale, hh = y3 * scale;
  float l0 = y0 - 0.5f * (ww - 1.0f), l1 = y1 - 0.5f * (hh - 1.0f);
  float l2 = y0 + 0.5f * (ww - 1.0f), l3 = y1 + 0.5f * (hh - 1.0f);
  int row = p / 32, col = p - row * 32;
  float mx = col * 16.0f, my = row * 16.0f;
  anc[4 * i + 0] = fminf(fmaxf(l0 + mx, 0.0f), 512.0f);
  anc[4 * i + 1] = fminf(fmaxf(l1 + my, 0.0f), 512.0f);
  anc[4 * i + 2] = fminf(fmaxf(l2 + mx, 0.0f), 512.0f);
  anc[4 * i + 3] = fminf(fmaxf(l3 + my, 0.0f), 512.0f);
}

__global__ void iou_at_k(const float* __restrict__ bbx, const float* __restrict__ anc,
                         float* __restrict__ iou0, int* __restrict__ at)
{
  int i = blockIdx.x * 256 + threadIdx.x;
  if (i >= NA) return;
  float gx1 = bbx[0], gy1 = bbx[1], gx2 = bbx[2], gy2 = bbx[3];
  float ga = (gx2 - gx1) * (gy2 - gy1);
  float x1 = anc[4 * i], y1 = anc[4 * i + 1], x2 = anc[4 * i + 2], y2 = anc[4 * i + 3];
  float aa = (x2 - x1) * (y2 - y1);
  float lx = fmaxf(gx1, x1), ly = fmaxf(gy1, y1);
  float rx = fminf(gx2, x2), ry = fminf(gy2, y2);
  float iw = fmaxf(rx - lx, 0.0f), ih = fmaxf(ry - ly, 0.0f);
  float inter = iw * ih;
  float iou = inter / (ga + aa - inter);
  iou0[i] = iou;
  int lab = -1;
  if (iou >= 0.5f) lab = 1;
  if (iou <= 0.1f) lab = 0;
  at[i] = lab;
}

// ---------------- bitonic sort (global memory, stable via idx tie-break) ---
__global__ void sort_init1(const float* __restrict__ smax, float* key, int* idx)
{
  int i = blockIdx.x * 256 + threadIdx.x;
  if (i >= 16384) return;
  key[i] = (i < NA) ? smax[i] : 1e30f;   // ascending: pads go last
  idx[i] = i;
}

__global__ void sort_init2(const float* __restrict__ iou0, const int* __restrict__ sidx,
                           float* key, int* idx)
{
  int i = blockIdx.x * 256 + threadIdx.x;
  if (i >= 8192) return;
  key[i] = (i < TOPN) ? iou0[sidx[i]] : -1e30f;  // descending: pads go last
  idx[i] = i;
}

__global__ void bitonic_step(float* __restrict__ key, int* __restrict__ idx,
                             int j, int k, int n, int desc)
{
  int i = blockIdx.x * 256 + threadIdx.x;
  if (i >= n) return;
  int l = i ^ j;
  if (l <= i) return;
  float ki = key[i], kl = key[l];
  int ii = idx[i], il = idx[l];
  // "elem i should come after elem l" in the FINAL order
  bool after = desc ? (ki < kl || (ki == kl && ii > il))
                    : (ki > kl || (ki == kl && ii > il));
  bool dir = ((i & k) == 0);
  bool doswap = dir ? after : !after;
  if (doswap) { key[i] = kl; key[l] = ki; idx[i] = il; idx[l] = ii; }
}

__global__ void boxes_sorted_k(const int* __restrict__ order, const int* __restrict__ sidx,
                               const float* __restrict__ anc, float* __restrict__ bs)
{
  int r = blockIdx.x * 256 + threadIdx.x;
  if (r >= TOPN) return;
  int ti = sidx[order[r]];
  bs[4 * r + 0] = anc[4 * ti + 0];
  bs[4 * r + 1] = anc[4 * ti + 1];
  bs[4 * r + 2] = anc[4 * ti + 2];
  bs[4 * r + 3] = anc[4 * ti + 3];
}

__global__ void nms_supmat(const float* __restrict__ bs, unsigned* __restrict__ sup)
{
  int t = blockIdx.x * 256 + threadIdx.x;
  if (t >= TOPN * NW) return;
  int i = t / NW, wdx = t - (t / NW) * NW;
  float ax1 = bs[4 * i], ay1 = bs[4 * i + 1], ax2 = bs[4 * i + 2], ay2 = bs[4 * i + 3];
  float areaA = (ax2 - ax1) * (ay2 - ay1);
  unsigned m = 0;
  int jb = wdx * 32;
  for (int bte = 0; bte < 32; ++bte) {
    int j = jb + bte;
    if (j >= TOPN) break;
    float bx1 = bs[4 * j], by1 = bs[4 * j + 1], bx2 = bs[4 * j + 2], by2 = bs[4 * j + 3];
    float areaB = (bx2 - bx1) * (by2 - by1);
    float lx = fmaxf(ax1, bx1), ly = fmaxf(ay1, by1);
    float rx = fminf(ax2, bx2), ry = fminf(ay2, by2);
    float iw = fmaxf(rx - lx, 0.0f), ih = fmaxf(ry - ly, 0.0f);
    float inter = iw * ih;
    float iou = inter / (areaA + areaB - inter);
    if (iou > 0.6f) m |= (1u << bte);
  }
  sup[(size_t)i * NW + wdx] = m;
}

// blocked greedy NMS scan: 188 outer iterations, 2 barriers each
__global__ void nms_scan(const unsigned* __restrict__ sup, unsigned* __restrict__ keepg)
{
  __shared__ unsigned keep[NW];
  int tid = threadIdx.x;
  for (int w = tid; w < NW; w += 256) keep[w] = (w == NW - 1) ? 0x0000FFFFu : 0xFFFFFFFFu;
  __syncthreads();
  for (int b = 0; b < NW; ++b) {
    if (tid == 0) {
      unsigned kw = keep[b];
      for (int bit = 0; bit < 32; ++bit) {
        int i = b * 32 + bit;
        if (i >= TOPN) break;
        if ((kw >> bit) & 1u) {
          unsigned m = sup[(size_t)i * NW + b];
          unsigned gt = (bit == 31) ? 0u : (0xFFFFFFFFu << (bit + 1));
          kw &= ~(m & gt);
        }
      }
      keep[b] = kw;
    }
    __syncthreads();
    unsigned kw = keep[b];
    for (int w = b + 1 + tid; w < NW; w += 256) {
      unsigned acc = 0, t = kw;
      while (t) {
        int bit = __ffs(t) - 1;
        t &= t - 1;
        acc |= sup[(size_t)(b * 32 + bit) * NW + w];
      }
      keep[w] &= ~acc;
    }
    __syncthreads();
  }
  for (int w = tid; w < NW; w += 256) keepg[w] = keep[w];
}

// replicate the reference bug: nms indices index the FULL anchor set
__global__ void nms_collect(const unsigned* __restrict__ keepg, const int* __restrict__ order,
                            const float* __restrict__ anc, const int* __restrict__ at,
                            float* __restrict__ roi, int* __restrict__ roit)
{
  if (threadIdx.x != 0 || blockIdx.x != 0) return;
  int cnt = 0;
  for (int r = 0; r < TOPN && cnt < KROI; ++r) {
    if ((keepg[r >> 5] >> (r & 31)) & 1u) {
      int idx = order[r];
      roi[4 * cnt + 0] = anc[4 * idx + 0];
      roi[4 * cnt + 1] = anc[4 * idx + 1];
      roi[4 * cnt + 2] = anc[4 * idx + 2];
      roi[4 * cnt + 3] = anc[4 * idx + 3];
      roit[cnt] = at[idx];
      ++cnt;
    }
  }
  int idx0 = order[0];
  for (; cnt < KROI; ++cnt) {
    roi[4 * cnt + 0] = anc[4 * idx0 + 0];
    roi[4 * cnt + 1] = anc[4 * idx0 + 1];
    roi[4 * cnt + 2] = anc[4 * idx0 + 2];
    roi[4 * cnt + 3] = anc[4 * idx0 + 3];
    roit[cnt] = at[idx0];
  }
}

// torchvision roi_pool semantics, spatial_scale = 1.0 on a 32x32 map
__global__ __launch_bounds__(512) void roi_pool_k(
    const float* __restrict__ feat, const float* __restrict__ roi,
    float* __restrict__ pooled)
{
  int k = blockIdx.x;       // 0..127
  int c = threadIdx.x;      // 0..511
  int sw = (int)rintf(roi[4 * k + 0]);
  int sh = (int)rintf(roi[4 * k + 1]);
  int ew = (int)rintf(roi[4 * k + 2]);
  int eh = (int)rintf(roi[4 * k + 3]);
  float bw = (float)max(ew - sw + 1, 1) / 7.0f;
  float bh = (float)max(eh - sh + 1, 1) / 7.0f;
  const float* f = feat + (size_t)c * FH * FW;
  for (int ph = 0; ph < 7; ++ph) {
    int hs = min(max((int)floorf(ph * bh) + sh, 0), FH);
    int he = min(max((int)ceilf((ph + 1) * bh) + sh, 0), FH);
    for (int pw = 0; pw < 7; ++pw) {
      int ws = min(max((int)floorf(pw * bw) + sw, 0), FW);
      int we = min(max((int)ceilf((pw + 1) * bw) + sw, 0), FW);
      float m = 0.0f;
      if (he > hs && we > ws) {
        m = -3.4e38f;
        for (int h = hs; h < he; ++h)
          for (int w = ws; w < we; ++w)
            m = fmaxf(m, f[h * FW + w]);
      }
      pooled[(size_t)k * 25088 + (size_t)c * 49 + ph * 7 + pw] = m;
    }
  }
}

// ---------------------------------------------------------------------------
// FC GEMM: D = A(MxK,f32) * B(KxN,f32) + bias, on V_WMMA_F32_16X16X32_F16.
// M == 128: the 8 waves of a workgroup are the 8 M-tiles of one N-tile, and
// they share one B tile, staged cooperatively into LDS (f32->f16, transposed
// to [n][k] so each lane's fragment is two contiguous 16B ds loads).  A
// fragments are four contiguous float4 (b128) loads.  B (the 411 MB fc1
// weight stream) is fetched from HBM exactly once per workgroup.
// ---------------------------------------------------------------------------
__global__ __launch_bounds__(256) void gemm_f16wmma(
    const float* __restrict__ A, const float* __restrict__ B,
    const float* __restrict__ bias, float* __restrict__ out,
    int M, int N, int K, int relu)
{
  __shared__ _Float16 bt[16][32];            // [n][k] tile, 1 KB
  const int tid = threadIdx.x;
  const int wave = tid >> 5;
  const int lane = tid & 31;
  const int lr = lane & 15, half = lane >> 4;
  const int m0 = wave * 16;
  const int n0 = blockIdx.x * 16;
  const int nr = tid & 15, kr = tid >> 4;    // staging coords (16x16 threads)
  union { v16h v; _Float16 h[16]; } a;
  union { v16h v; v8h g[2]; } b;
  union { v8f v; float f[8]; } acc;
  #pragma unroll
  for (int r = 0; r < 8; ++r) acc.f[r] = 0.0f;
  const float* Ap  = A + (size_t)(m0 + lr) * K + half * 8;
  const float* Bs0 = B + (size_t)kr * N + n0 + nr;
  const float* Bs1 = B + (size_t)(kr + 16) * N + n0 + nr;
  for (int k0 = 0; k0 < K; k0 += 32) {
    __syncthreads();
    // cooperative stage of the shared B tile (converted + transposed)
    bt[nr][kr]      = (_Float16)Bs0[0];
    bt[nr][kr + 16] = (_Float16)Bs1[0];
    if (k0 + 96 < K) __builtin_prefetch(Bs0 + (size_t)96 * N, 0, 1);
    Bs0 += (size_t)32 * N;
    Bs1 += (size_t)32 * N;
    __syncthreads();
    // A fragment: K = k0 + {0..7,16..23} + 8*half  (four b128 loads)
    const float4 x0 = *(const float4*)(Ap + 0);
    const float4 x1 = *(const float4*)(Ap + 4);
    const float4 x2 = *(const float4*)(Ap + 16);
    const float4 x3 = *(const float4*)(Ap + 20);
    Ap += 32;
    a.h[0]  = (_Float16)x0.x; a.h[1]  = (_Float16)x0.y;
    a.h[2]  = (_Float16)x0.z; a.h[3]  = (_Float16)x0.w;
    a.h[4]  = (_Float16)x1.x; a.h[5]  = (_Float16)x1.y;
    a.h[6]  = (_Float16)x1.z; a.h[7]  = (_Float16)x1.w;
    a.h[8]  = (_Float16)x2.x; a.h[9]  = (_Float16)x2.y;
    a.h[10] = (_Float16)x2.z; a.h[11] = (_Float16)x2.w;
    a.h[12] = (_Float16)x3.x; a.h[13] = (_Float16)x3.y;
    a.h[14] = (_Float16)x3.z; a.h[15] = (_Float16)x3.w;
    // B fragment from LDS: k = {0..7,16..23} + 8*half at n = n0+lr
    const v8h* bp = (const v8h*)&bt[lr][half * 8];
    b.g[0] = bp[0];                          // k = 8h + 0..7
    b.g[1] = bp[2];                          // k = 16 + 8h + 0..7
    acc.v = __builtin_amdgcn_wmma_f32_16x16x32_f16(false, a.v, false, b.v,
                                                   (short)0, acc.v, false, false);
  }
  #pragma unroll
  for (int r = 0; r < 8; ++r) {
    int m = m0 + r + (half ? 8 : 0);
    int n = n0 + lr;
    float v = acc.f[r] + bias[n];
    if (relu) v = fmaxf(v, 0.0f);
    out[(size_t)m * N + n] = v;
  }
}

__global__ void head_gemm(const float* __restrict__ A, const float* __restrict__ W,
                          const float* __restrict__ b, float* __restrict__ out,
                          int M, int N, int K)
{
  int t = blockIdx.x * 256 + threadIdx.x;
  if (t >= M * N) return;
  int m = t / N, n = t - m * N;
  float s = 0.0f;
  for (int k = 0; k < K; ++k) s += A[(size_t)m * K + k] * W[(size_t)k * N + n];
  out[t] = s + b[n];
}

__global__ __launch_bounds__(1024) void loss_rpn_k(
    const float* __restrict__ prop, const float* __restrict__ anc,
    const float* __restrict__ sc2, const int* __restrict__ at, float* __restrict__ outv)
{
  __shared__ float sA[1024], sB[1024], sC[1024], sD[1024];
  int tid = threadIdx.x;
  float bx = 0.0f, pcnt = 0.0f, nll = 0.0f, vcnt = 0.0f;
  for (int i = tid; i < NA; i += 1024) {
    int lab = at[i];
    if (lab > 0) {
      pcnt += 1.0f;
      for (int j = 0; j < 4; ++j) {
        float d = fabsf(prop[4 * i + j] - anc[4 * i + j]);
        bx += (d < 1.0f) ? 0.5f * d * d : d - 0.5f;
      }
    }
    if (lab >= 0) {
      float p0 = sc2[2 * i], p1 = sc2[2 * i + 1];
      float mx = fmaxf(p0, p1);
      float ls = mx + logf(expf(p0 - mx) + expf(p1 - mx));  // log_softmax of probs
      float q = ((lab == 1) ? p1 : p0) - ls;
      nll += -q;
      vcnt += 1.0f;
    }
  }
  sA[tid] = bx; sB[tid] = pcnt; sC[tid] = nll; sD[tid] = vcnt;
  __syncthreads();
  for (int s = 512; s > 0; s >>= 1) {
    if (tid < s) { sA[tid] += sA[tid+s]; sB[tid] += sB[tid+s]; sC[tid] += sC[tid+s]; sD[tid] += sD[tid+s]; }
    __syncthreads();
  }
  if (tid == 0) {
    outv[0] = sA[0] / (4.0f * sB[0]);
    outv[1] = sC[0] / sD[0];
  }
}

__global__ __launch_bounds__(128) void loss_roi_k(
    const float* __restrict__ bxo, const float* __restrict__ roi,
    const float* __restrict__ clsl, const int* __restrict__ roit, float* __restrict__ outv)
{
  __shared__ float sA[128], sC[128], sD[128];
  int tid = threadIdx.x;        // 0..127 == roi index
  float bxs = 0.0f, nll = 0.0f, vcnt = 0.0f;
  for (int j = 0; j < 4; ++j) {
    float d = fabsf(bxo[4 * tid + j] - roi[4 * tid + j]);
    bxs += (d < 1.0f) ? 0.5f * d * d : d - 0.5f;
  }
  float l0 = clsl[2 * tid], l1 = clsl[2 * tid + 1];
  float mx = fmaxf(l0, l1);
  float e0 = expf(l0 - mx), e1 = expf(l1 - mx);
  float s = e0 + e1;
  float p0 = e0 / s, p1 = e1 / s;                  // softmax
  float mx2 = fmaxf(p0, p1);
  float ls = mx2 + logf(expf(p0 - mx2) + expf(p1 - mx2));  // log_softmax again
  int lab = roit[tid];
  if (lab >= 0) { nll = -(((lab == 1) ? p1 : p0) - ls); vcnt = 1.0f; }
  sA[tid] = bxs; sC[tid] = nll; sD[tid] = vcnt;
  __syncthreads();
  for (int st = 64; st > 0; st >>= 1) {
    if (tid < st) { sA[tid] += sA[tid+st]; sC[tid] += sC[tid+st]; sD[tid] += sD[tid+st]; }
    __syncthreads();
  }
  if (tid == 0) {
    outv[2] = sA[0] / 512.0f;
    outv[3] = sC[0] / sD[0];
  }
}

// ---------------------------------------------------------------------------
extern "C" void kernel_launch(void* const* d_in, const int* in_sizes, int n_in,
                              void* d_out, int out_size, void* d_ws, size_t ws_size,
                              hipStream_t stream) {
  (void)in_sizes; (void)n_in; (void)out_size; (void)ws_size;
  const float* image = (const float*)d_in[0];
  const float* bbx   = (const float*)d_in[1];
  const float* vggw[13]; const float* vggb[13];
  for (int i = 0; i < 13; ++i) { vggw[i] = (const float*)d_in[3 + 2*i]; vggb[i] = (const float*)d_in[4 + 2*i]; }
  const float* rpnc_w = (const float*)d_in[29]; const float* rpnc_b = (const float*)d_in[30];
  const float* cls1_w = (const float*)d_in[31]; const float* cls1_b = (const float*)d_in[32];
  const float* bbx1_w = (const float*)d_in[33]; const float* bbx1_b = (const float*)d_in[34];
  const float* fc1_w  = (const float*)d_in[35]; const float* fc1_b  = (const float*)d_in[36];
  const float* fc2_w  = (const float*)d_in[37]; const float* fc2_b  = (const float*)d_in[38];
  const float* clsh_w = (const float*)d_in[39]; const float* clsh_b = (const float*)d_in[40];
  const float* bxh_w  = (const float*)d_in[41]; const float* bxh_b  = (const float*)d_in[42];
  float* outv = (float*)d_out;

  char* base = (char*)d_ws;
  size_t off = 0;
  auto alloc = [&](size_t n) { void* p = (void*)(base + off); off += (n + 255) & ~(size_t)255; return p; };
  float*    bufA   = (float*)alloc((size_t)64 * 512 * 512 * 4);
  float*    bufB   = (float*)alloc((size_t)64 * 512 * 512 * 4);
  float*    rpnx   = (float*)alloc((size_t)512 * 1024 * 4);
  float*    cls18  = (float*)alloc((size_t)18 * 1024 * 4);
  float*    bbx36  = (float*)alloc((size_t)36 * 1024 * 4);
  float*    sc2    = (float*)alloc((size_t)NA * 2 * 4);
  float*    smax   = (float*)alloc((size_t)NA * 4);
  float*    prop   = (float*)alloc((size_t)NA * 4 * 4);
  float*    anc    = (float*)alloc((size_t)NA * 4 * 4);
  float*    iou0   = (float*)alloc((size_t)NA * 4);
  int*      at     = (int*)  alloc((size_t)NA * 4);
  float*    skey   = (float*)alloc((size_t)16384 * 4);
  int*      sidx   = (int*)  alloc((size_t)16384 * 4);
  float*    key2   = (float*)alloc((size_t)8192 * 4);
  int*      idx2   = (int*)  alloc((size_t)8192 * 4);
  float*    bsort  = (float*)alloc((size_t)TOPN * 4 * 4);
  unsigned* sup    = (unsigned*)alloc((size_t)TOPN * NW * 4);
  unsigned* keepg  = (unsigned*)alloc((size_t)NW * 4);
  float*    roi    = (float*)alloc((size_t)KROI * 4 * 4);
  int*      roit   = (int*)  alloc((size_t)KROI * 4);
  float*    pooled = (float*)alloc((size_t)KROI * 25088 * 4);
  float*    h1     = (float*)alloc((size_t)KROI * 4096 * 4);
  float*    h2     = (float*)alloc((size_t)KROI * 4096 * 4);
  float*    clsl   = (float*)alloc((size_t)KROI * 2 * 4);
  float*    bxo    = (float*)alloc((size_t)KROI * 4 * 4);

  // ---- VGG feature stack ----
  static const int cinA[13]  = {3,64,64,128,128,256,256,256,512,512,512,512,512};
  static const int coutA[13] = {64,64,128,128,256,256,256,512,512,512,512,512,512};
  static const int poolAfter[13] = {0,1,0,1,0,0,1,0,0,1,0,0,0};
  int H = 512, W = 512;
  const float* cur = image;
  float* dst = bufA;
  for (int i = 0; i < 13; ++i) {
    dim3 g((H * W) / 128, coutA[i] / 16);
    conv3x3_wmma<<<g, 256, 0, stream>>>(cur, vggw[i], vggb[i], dst,
                                        cinA[i], coutA[i], H, W, 1);
    cur = dst; dst = (dst == bufA) ? bufB : bufA;
    if (poolAfter[i]) {
      int n = coutA[i] * (H / 2) * (W / 2);
      maxpool2_k<<<(n + 255) / 256, 256, 0, stream>>>(cur, dst, coutA[i], H, W);
      cur = dst; dst = (dst == bufA) ? bufB : bufA;
      H >>= 1; W >>= 1;
    }
  }
  const float* feat = cur;   // (512, 32, 32)

  // ---- RPN ----
  {
    dim3 g(1024 / 128, 512 / 16);
    conv3x3_wmma<<<g, 256, 0, stream>>>(feat, rpnc_w, rpnc_b, rpnx, 512, 512, 32, 32, 1);
  }
  conv1x1_k<<<(18 * 1024 + 255) / 256, 256, 0, stream>>>(rpnx, cls1_w, cls1_b, cls18, 18, 512, 1024);
  conv1x1_k<<<(36 * 1024 + 255) / 256, 256, 0, stream>>>(rpnx, bbx1_w, bbx1_b, bbx36, 36, 512, 1024);
  scores_k<<<(NA + 255) / 256, 256, 0, stream>>>(cls18, sc2, smax);
  proposals_k<<<(NA * 4 + 255) / 256, 256, 0, stream>>>(bbx36, prop);
  anchors_k<<<(NA + 255) / 256, 256, 0, stream>>>(anc);
  iou_at_k<<<(NA + 255) / 256, 256, 0, stream>>>(bbx, anc, iou0, at);

  // ---- top-N selection: ascending stable argsort of smax (reference bug) ----
  sort_init1<<<16384 / 256, 256, 0, stream>>>(smax, skey, sidx);
  for (int k = 2; k <= 16384; k <<= 1)
    for (int j = k >> 1; j > 0; j >>= 1)
      bitonic_step<<<16384 / 256, 256, 0, stream>>>(skey, sidx, j, k, 16384, 0);

  // ---- NMS ordering: descending stable argsort of IoU-with-GT ----
  sort_init2<<<8192 / 256, 256, 0, stream>>>(iou0, sidx, key2, idx2);
  for (int k = 2; k <= 8192; k <<= 1)
    for (int j = k >> 1; j > 0; j >>= 1)
      bitonic_step<<<8192 / 256, 256, 0, stream>>>(key2, idx2, j, k, 8192, 1);

  boxes_sorted_k<<<(TOPN + 255) / 256, 256, 0, stream>>>(idx2, sidx, anc, bsort);
  nms_supmat<<<(TOPN * NW + 255) / 256, 256, 0, stream>>>(bsort, sup);
  nms_scan<<<1, 256, 0, stream>>>(sup, keepg);
  nms_collect<<<1, 32, 0, stream>>>(keepg, idx2, anc, at, roi, roit);

  // ---- ROI pool + FC head ----
  roi_pool_k<<<KROI, 512, 0, stream>>>(feat, roi, pooled);
  gemm_f16wmma<<<4096 / 16, 256, 0, stream>>>(pooled, fc1_w, fc1_b, h1, KROI, 4096, 25088, 1);
  gemm_f16wmma<<<4096 / 16, 256, 0, stream>>>(h1, fc2_w, fc2_b, h2, KROI, 4096, 4096, 1);
  head_gemm<<<(KROI * 2 + 255) / 256, 256, 0, stream>>>(h2, clsh_w, clsh_b, clsl, KROI, 2, 4096);
  head_gemm<<<(KROI * 4 + 255) / 256, 256, 0, stream>>>(h2, bxh_w, bxh_b, bxo, KROI, 4, 4096);

  // ---- losses ----
  loss_rpn_k<<<1, 1024, 0, stream>>>(prop, anc, sc2, at, outv);
  loss_roi_k<<<1, 128, 0, stream>>>(bxo, roi, clsl, roit, outv);
}
